// ScaledDotSelfAttention_76364518523526
// MI455X (gfx1250) — matrix operand order repaired and checked by reading, not compiled, not run on hardware
//
#include <hip/hip_runtime.h>

// ---------------------------------------------------------------------------
// CDNA5 (gfx1250) wave32 WMMA self-attention.
//   B=2, T=2048, D=1024, H=16, U=64, OUT=1024
// Pipeline:
//   k_mask_convert      : x = inputs * mask  -> bf16 [B,T,D]
//   k_transpose_convert : W (f32 [K,N]) -> Wt (bf16 [N,K])  (x4 weights)
//   k_gemm16 mode0      : Q,K = x @ W   -> bf16 [B,H,T,U]
//   k_gemm16 mode1      : V   = x @ Wv  -> bf16 [B,H,U,T]   (pre-transposed)
//   k_attn              : flash attention w/ masked-renorm softmax -> bf16 [B,T,H*U]
//   k_gemm16 mode2      : out = attended @ Wo -> f32 [B,T,1024]
// GEMM: 4 waves/WG, 64x64 tile, B staged via double-buffered
// global_load_async_to_lds_b128 (ASYNCcnt) overlapping the WMMA stream.
// All WMMA operands are pure SSA u32x4 values (no loop-carried unions) so
// nothing spills to scratch across the inline-asm "memory" clobbers.
// ---------------------------------------------------------------------------

typedef __bf16        v16bf __attribute__((ext_vector_type(16)));
typedef float         v8f   __attribute__((ext_vector_type(8)));
typedef unsigned int  u32x4 __attribute__((ext_vector_type(4)));

__device__ __forceinline__ v16bf mkv(u32x4 lo, u32x4 hi) {
  union U { u32x4 q[2]; v16bf v; } u;
  u.q[0] = lo; u.q[1] = hi;
  return u.v;
}

__device__ __forceinline__ u32x4 ld4(const __bf16* p) {   // 16B load (global or LDS)
  return *(const u32x4*)p;
}

__device__ __forceinline__ v8f wmma_bf16(v16bf a, v16bf b, v8f c) {
  // D = A(16x32 bf16) * B(32x16 bf16) + C(16x16 f32)
  return __builtin_amdgcn_wmma_f32_16x16x32_bf16(
      /*neg_a=*/false, a, /*neg_b=*/false, b,
      /*c_mod=*/(short)0, c, /*reuse_a=*/false, /*reuse_b=*/false);
}

// ---------------------------------------------------------------------------
__global__ __launch_bounds__(256)
void k_mask_convert(const float* __restrict__ x, const unsigned char* __restrict__ mask,
                    __bf16* __restrict__ xb, int D, long long total) {
  long long i = (long long)blockIdx.x * 256 + threadIdx.x;
  if (i >= total) return;
  float m = mask[i / D] ? 1.0f : 0.0f;
  xb[i] = (__bf16)(x[i] * m);
}

__global__ __launch_bounds__(256)
void k_transpose_convert(const float* __restrict__ W, __bf16* __restrict__ Wt,
                         int K, int N) {
  long long i = (long long)blockIdx.x * 256 + threadIdx.x;
  if (i >= (long long)K * N) return;
  int k = (int)(i / N), n = (int)(i % N);
  Wt[(size_t)n * K + k] = (__bf16)W[i];
}

// ---------------------------------------------------------------------------
// Async fill of one 64-row x 32-col bf16 B tile into LDS (padded row stride 40).
// 256 x 16B chunks, 2 per thread (128 threads) -> 2 async instrs per wave.
__device__ __forceinline__ void async_fill_tile(const __bf16* __restrict__ Bt,
                                                __bf16* lds, int Kdim, int n0,
                                                int kt, int tid) {
#pragma unroll
  for (int it = 0; it < 2; ++it) {
    int c = tid + it * 128;
    int row = c >> 2, part = c & 3;
    const __bf16* g = Bt + (size_t)(n0 + row) * Kdim + kt + part * 8;
    unsigned ldsOff =
        (unsigned)(unsigned long long)(lds + row * 40 + part * 8);
    asm volatile("global_load_async_to_lds_b128 %0, %1, off"
                 :: "v"(ldsOff), "v"((unsigned long long)g)
                 : "memory");
  }
}

// Generic bf16 GEMM: 4 waves/WG, 64x64 tile, K-step 32, double-buffered LDS B.
// A row-major [M,lda]; Bt = B^T row-major [N,Kdim].
// mode 0: out bf16 [bz][H][T][U]  (Q, K)
// mode 1: out bf16 [bz][H][U][T]  (V transposed)
// mode 2: out f32  row-major [M,1024]
__global__ __launch_bounds__(128)
void k_gemm16(const __bf16* __restrict__ A, const __bf16* __restrict__ Bt,
              __bf16* __restrict__ outb, float* __restrict__ outf,
              int Kdim, int lda, long long aStride, int Trows, int mode) {
  const int tid = threadIdx.x;
  const int w = tid >> 5, lane = tid & 31;
  const int hi = lane >> 4, lm = lane & 15;
  const int m0 = blockIdx.x * 64, n0 = blockIdx.y * 64;
  const int bz = blockIdx.z;
  const int cA = hi * 16;

  __shared__ __align__(16) __bf16 Bs[2][64 * 40];

  const __bf16* Arow =
      A + (size_t)bz * aStride + (size_t)(m0 + w * 16 + lm) * lda;

  // Prologue: fill buffer 0 (kt=0), then buffer 1 (kt=32) in flight.
  async_fill_tile(Bt, &Bs[0][0], Kdim, n0, 0, tid);
  asm volatile("s_wait_asynccnt 0x0" ::: "memory");
  async_fill_tile(Bt, &Bs[1][0], Kdim, n0, 32, tid);
  __syncthreads();

  u32x4 a0 = ld4(Arow + cA);
  u32x4 a1 = ld4(Arow + cA + 8);

  v8f acc[4] = {};
  int cur = 0;
  for (int kt = 0; kt < Kdim; kt += 32) {
    // software-pipeline next A operand (uniform integer select keeps it a
    // global_load; last iteration harmlessly reloads the current chunk)
    int ktn = (kt + 32 < Kdim) ? (kt + 32) : kt;
    u32x4 na0 = ld4(Arow + ktn + cA);
    u32x4 na1 = ld4(Arow + ktn + cA + 8);

    // issue all 8 B-operand LDS reads, then the 4 WMMAs
    const __bf16* bs = &Bs[cur][0] + lm * 40 + hi * 16;
    u32x4 b00 = ld4(bs +    0), b01 = ld4(bs +    8);
    u32x4 b10 = ld4(bs +  640), b11 = ld4(bs +  648);
    u32x4 b20 = ld4(bs + 1280), b21 = ld4(bs + 1288);
    u32x4 b30 = ld4(bs + 1920), b31 = ld4(bs + 1928);

    v16bf av = mkv(a0, a1);
    acc[0] = wmma_bf16(av, mkv(b00, b01), acc[0]);
    acc[1] = wmma_bf16(av, mkv(b10, b11), acc[1]);
    acc[2] = wmma_bf16(av, mkv(b20, b21), acc[2]);
    acc[3] = wmma_bf16(av, mkv(b30, b31), acc[3]);

    __syncthreads();  // everyone done reading Bs[cur]
    if (kt + 64 < Kdim) {
      async_fill_tile(Bt, &Bs[cur][0], Kdim, n0, kt + 64, tid);
      asm volatile("s_wait_asynccnt 0x2" ::: "memory");  // buf[nxt] ready
    } else {
      asm volatile("s_wait_asynccnt 0x0" ::: "memory");
    }
    __syncthreads();
    cur ^= 1;
    a0 = na0; a1 = na1;
  }

#pragma unroll
  for (int j = 0; j < 4; ++j) {
#pragma unroll
    for (int v = 0; v < 8; ++v) {
      int m = m0 + w * 16 + v + hi * 8;  // C layout: M = vgpr + 8*(lane>=16)
      int n = n0 + j * 16 + lm;          //           N = lane % 16
      float val = acc[j][v];
      if (mode == 0) {
        outb[(((size_t)bz * 16 + (n >> 6)) * Trows + m) * 64 + (n & 63)] = (__bf16)val;
      } else if (mode == 1) {
        outb[(((size_t)bz * 16 + (n >> 6)) * 64 + (n & 63)) * Trows + m] = (__bf16)val;
      } else {
        outf[(size_t)m * 1024 + n] = val;
      }
    }
  }
}

// ---------------------------------------------------------------------------
// Flash attention, 1 wave handles 16 query rows of one (b,h).
// weights = e^{l-max} * m / (Zm + eps*Zf)  ==  softmax*m / (sum(softmax*m)+eps)
__global__ __launch_bounds__(32)
void k_attn(const __bf16* __restrict__ q, const __bf16* __restrict__ kmat,
            const __bf16* __restrict__ vt, const unsigned char* __restrict__ mask,
            __bf16* __restrict__ att, int T) {
  const int lane = threadIdx.x, hi = lane >> 4, lm = lane & 15;
  const int bh = blockIdx.y, b = bh >> 4, h = bh & 15;
  const int m0 = blockIdx.x * 16;
  const int cA = hi * 16;

  __shared__ __align__(16) __bf16 pT[16 * 40];  // 16x32 P tile, row stride 40

  // Q A-operands (row = m0+lm, K-dim u: tiles [0,32) and [32,64)), SSA values
  const __bf16* qrow = q + ((size_t)bh * T + m0 + lm) * 64;
  u32x4 aq00 = ld4(qrow + cA),      aq01 = ld4(qrow + cA + 8);
  u32x4 aq10 = ld4(qrow + 32 + cA), aq11 = ld4(qrow + 32 + cA + 8);

  v8f O[4] = {};                 // 16x64 output accumulator (4 N-tiles of u)
  float rmax[8], zf[8], zm[8];   // per-row stats, row = v + 8*hi
#pragma unroll
  for (int v = 0; v < 8; ++v) { rmax[v] = -3.0e38f; zf[v] = 0.f; zm[v] = 0.f; }

  const __bf16* kb = kmat + (size_t)bh * T * 64;
  const __bf16* vb = vt + (size_t)bh * 64 * T;
  const unsigned char* mrow = mask + (size_t)b * T;
  const float scale = 0.125f;  // 1/sqrt(64)

  for (int s0 = 0; s0 < T; s0 += 32) {
    // warm WGP$ for the next S-tile while this one computes
    if (s0 + 32 < T) {
      __builtin_prefetch(kb + (size_t)(s0 + 32 + lm) * 64 + hi * 16);
      __builtin_prefetch(kb + (size_t)(s0 + 48 + lm) * 64 + hi * 16);
    }

    // ---- logits: load all 8 K chunks, then 4 WMMAs ----
    const __bf16* kr0 = kb + (size_t)(s0 + lm) * 64 + hi * 16;
    const __bf16* kr1 = kr0 + 16 * 64;
    u32x4 k00 = ld4(kr0),      k01 = ld4(kr0 + 8);
    u32x4 k02 = ld4(kr0 + 32), k03 = ld4(kr0 + 40);
    u32x4 k10 = ld4(kr1),      k11 = ld4(kr1 + 8);
    u32x4 k12 = ld4(kr1 + 32), k13 = ld4(kr1 + 40);

    v16bf aqv0 = mkv(aq00, aq01), aqv1 = mkv(aq10, aq11);
    v8f l0 = {}, l1 = {};
    l0 = wmma_bf16(aqv0, mkv(k00, k01), l0);
    l0 = wmma_bf16(aqv1, mkv(k02, k03), l0);
    l1 = wmma_bf16(aqv0, mkv(k10, k11), l1);
    l1 = wmma_bf16(aqv1, mkv(k12, k13), l1);

    float msk0 = mrow[s0 + lm] ? 1.f : 0.f;
    float msk1 = mrow[s0 + 16 + lm] ? 1.f : 0.f;

    float p0[8], p1[8];
#pragma unroll
    for (int v = 0; v < 8; ++v) {
      float a0 = l0[v] * scale, a1 = l1[v] * scale;
      // row max over this iteration's 32 s-values (16-lane group reduce)
      float tv = fmaxf(a0, a1);
#pragma unroll
      for (int off = 1; off < 16; off <<= 1)
        tv = fmaxf(tv, __shfl_xor(tv, off, 32));
      float nm = fmaxf(rmax[v], tv);
      float alpha = __expf(rmax[v] - nm);
      rmax[v] = nm;
      float e0 = __expf(a0 - nm), e1 = __expf(a1 - nm);
      float q0 = e0 * msk0, q1 = e1 * msk1;
      float sf = e0 + e1, sm = q0 + q1;
#pragma unroll
      for (int off = 1; off < 16; off <<= 1) {
        sf += __shfl_xor(sf, off, 32);
        sm += __shfl_xor(sm, off, 32);
      }
      zf[v] = zf[v] * alpha + sf;
      zm[v] = zm[v] * alpha + sm;
      p0[v] = q0; p1[v] = q1;
#pragma unroll
      for (int j = 0; j < 4; ++j) O[j][v] *= alpha;
    }

    // ---- C-layout -> A-layout transpose of masked P via LDS ----
    __syncthreads();
#pragma unroll
    for (int v = 0; v < 8; ++v) {
      pT[(v + hi * 8) * 40 + lm]      = (__bf16)p0[v];
      pT[(v + hi * 8) * 40 + 16 + lm] = (__bf16)p1[v];
    }
    __syncthreads();
    u32x4 pa0 = ld4(&pT[lm * 40 + cA]);
    u32x4 pa1 = ld4(&pT[lm * 40 + cA + 8]);
    v16bf aPv = mkv(pa0, pa1);

    // ---- O += P @ V : load all 8 V chunks, then 4 WMMAs ----
    const __bf16* vr0 = vb + (size_t)(lm)      * T + s0 + hi * 16;
    const __bf16* vr1 = vb + (size_t)(16 + lm) * T + s0 + hi * 16;
    const __bf16* vr2 = vb + (size_t)(32 + lm) * T + s0 + hi * 16;
    const __bf16* vr3 = vb + (size_t)(48 + lm) * T + s0 + hi * 16;
    u32x4 v00 = ld4(vr0), v01 = ld4(vr0 + 8);
    u32x4 v10 = ld4(vr1), v11 = ld4(vr1 + 8);
    u32x4 v20 = ld4(vr2), v21 = ld4(vr2 + 8);
    u32x4 v30 = ld4(vr3), v31 = ld4(vr3 + 8);

    O[0] = wmma_bf16(aPv, mkv(v00, v01), O[0]);
    O[1] = wmma_bf16(aPv, mkv(v10, v11), O[1]);
    O[2] = wmma_bf16(aPv, mkv(v20, v21), O[2]);
    O[3] = wmma_bf16(aPv, mkv(v30, v31), O[3]);
  }

  // normalize: 1/(Zm + eps*Zf), store attended bf16 [B,T, h*64+u]
  float r[8];
#pragma unroll
  for (int v = 0; v < 8; ++v) r[v] = 1.0f / (zm[v] + 1e-7f * zf[v]);
#pragma unroll
  for (int j = 0; j < 4; ++j) {
#pragma unroll
    for (int v = 0; v < 8; ++v) {
      size_t idx = ((size_t)b * T + m0 + v + hi * 8) * 1024 + h * 64 + j * 16 + lm;
      att[idx] = (__bf16)(O[j][v] * r[v]);
    }
  }
}

// ---------------------------------------------------------------------------
extern "C" void kernel_launch(void* const* d_in, const int* in_sizes, int n_in,
                              void* d_out, int out_size, void* d_ws, size_t ws_size,
                              hipStream_t stream) {
  const float*         inputs = (const float*)d_in[0];
  const unsigned char* mask   = (const unsigned char*)d_in[1];
  const float*         Wq     = (const float*)d_in[2];
  const float*         Wk     = (const float*)d_in[3];
  const float*         Wv     = (const float*)d_in[4];
  const float*         Wo     = (const float*)d_in[5];
  float*               out    = (float*)d_out;

  const int B = 2, T = 2048, D = 1024, H = 16, HU = 1024;

  char* p = (char*)d_ws;
  auto take = [&](size_t bytes) -> char* {
    char* r = p; p += (bytes + 255) & ~(size_t)255; return r;
  };
  __bf16* xb   = (__bf16*)take((size_t)B * T * D * 2);
  __bf16* wqT  = (__bf16*)take((size_t)D * HU * 2);
  __bf16* wkT  = (__bf16*)take((size_t)D * HU * 2);
  __bf16* wvT  = (__bf16*)take((size_t)D * HU * 2);
  __bf16* woT  = (__bf16*)take((size_t)HU * 1024 * 2);
  __bf16* qb   = (__bf16*)take((size_t)B * H * T * 64 * 2);
  __bf16* kbuf = (__bf16*)take((size_t)B * H * T * 64 * 2);
  __bf16* vtb  = (__bf16*)take((size_t)B * H * 64 * T * 2);
  __bf16* attb = (__bf16*)take((size_t)B * T * HU * 2);

  long long totalX = (long long)B * T * D;
  k_mask_convert<<<(unsigned)((totalX + 255) / 256), 256, 0, stream>>>(
      inputs, mask, xb, D, totalX);

  int wElems = D * HU;
  k_transpose_convert<<<(wElems + 255) / 256, 256, 0, stream>>>(Wq, wqT, D, HU);
  k_transpose_convert<<<(wElems + 255) / 256, 256, 0, stream>>>(Wk, wkT, D, HU);
  k_transpose_convert<<<(wElems + 255) / 256, 256, 0, stream>>>(Wv, wvT, D, HU);
  k_transpose_convert<<<(wElems + 255) / 256, 256, 0, stream>>>(Wo, woT, HU, 1024);

  dim3 gproj(T / 64, HU / 64, B);
  k_gemm16<<<gproj, 128, 0, stream>>>(xb, wqT, qb,   nullptr, D, D, (long long)T * D, T, 0);
  k_gemm16<<<gproj, 128, 0, stream>>>(xb, wkT, kbuf, nullptr, D, D, (long long)T * D, T, 0);
  k_gemm16<<<gproj, 128, 0, stream>>>(xb, wvT, vtb,  nullptr, D, D, (long long)T * D, T, 1);

  dim3 gattn(T / 16, B * H, 1);
  k_attn<<<gattn, 32, 0, stream>>>(qb, kbuf, vtb, mask, attb, T);

  dim3 gout((B * T) / 64, 1024 / 64, 1);
  k_gemm16<<<gout, 128, 0, stream>>>(attb, woT, nullptr, out, HU, HU, 0ll, T, 2);
}
